// MiniGPTModel_30597347017106
// MI455X (gfx1250) — matrix operand order repaired and checked by reading, not compiled, not run on hardware
//
#include <hip/hip_runtime.h>
#include <hip/hip_bf16.h>
#include <math.h>

// ---------------------------------------------------------------------------
// MiniGPT forward on gfx1250 (CDNA5): bf16 WMMA GEMMs with fp32 accumulate.
// All weight matrices are pre-packed N-major [N,K] bf16 so the GEMM's LDS
// tiles need no transpose in the hot loop (ds_store_b128 only).
// ---------------------------------------------------------------------------

#define BB 16
#define TT 512
#define BT 8192       // BB*TT
#define EE 384
#define HH 8
#define DD 48
#define DP 64         // head dim padded to 64 for K=32 WMMA steps (Q/K only)
#define FF_ 1536
#define VV 20000
#define LL 8

typedef __bf16 bf16_t;
typedef bf16_t bf16x8  __attribute__((ext_vector_type(8)));
typedef bf16_t bf16x16 __attribute__((ext_vector_type(16)));
typedef float  f32x8   __attribute__((ext_vector_type(8)));

union Frag { bf16x16 v; bf16x8 h[2]; };

__device__ inline f32x8 zero8() {
  f32x8 z = {0.f, 0.f, 0.f, 0.f, 0.f, 0.f, 0.f, 0.f};
  return z;
}

__device__ inline bf16x8 zerob8() {
  bf16x8 z;
#pragma unroll
  for (int j = 0; j < 8; ++j) z[j] = (bf16_t)0.0f;
  return z;
}

__device__ inline f32x8 wmma_bf16(bf16x16 a, bf16x16 b, f32x8 c) {
  // D = A(16x32 bf16) * B(32x16 bf16) + C(16x16 f32)
  return __builtin_amdgcn_wmma_f32_16x16x32_bf16(false, a, false, b,
                                                 (short)0, c, false, false);
}

// ---------------------------------------------------------------------------
// Weight preparation kernels (fp32 -> bf16, packing, transposition to [N,K])
// ---------------------------------------------------------------------------

__global__ void k_cvt(const float* __restrict__ s, bf16_t* __restrict__ d, size_t n) {
  size_t i = (size_t)blockIdx.x * blockDim.x + threadIdx.x;
  if (i < n) d[i] = (bf16_t)s[i];
}

// Transpose-convert: src [L][K][N] fp32 -> dst [L][N][K] bf16
__global__ void k_cvtT(const float* __restrict__ s, bf16_t* __restrict__ d,
                       int K, int N, size_t total) {
  size_t i = (size_t)blockIdx.x * blockDim.x + threadIdx.x;
  if (i >= total) return;
  size_t kn = (size_t)K * N;
  int l = (int)(i / kn);
  int rem = (int)(i % kn);
  int n = rem / K, k = rem % K;
  d[i] = (bf16_t)s[(size_t)l * kn + (size_t)k * N + n];
}

// Pack Wq|Wk|Wv [L,E(k),384(n)] each -> N-major [L,1152,384] bf16
__global__ void k_pack_qkvw(const float* __restrict__ Wq, const float* __restrict__ Wk,
                            const float* __restrict__ Wv, bf16_t* __restrict__ dst) {
  size_t i = (size_t)blockIdx.x * blockDim.x + threadIdx.x;
  const size_t total = (size_t)LL * 1152 * EE;
  if (i >= total) return;
  int l   = (int)(i / ((size_t)1152 * EE));
  int rem = (int)(i % ((size_t)1152 * EE));
  int n = rem / EE, k = rem % EE;            // dst[l][n][k]
  int which = n / EE, c = n % EE;            // source column
  const float* W = (which == 0) ? Wq : ((which == 1) ? Wk : Wv);
  dst[i] = (bf16_t)W[((size_t)l * EE + k) * EE + c];
}

// Pack bq|bk|bv [L,384] each -> [L,1152] fp32
__global__ void k_pack_qkvb(const float* __restrict__ bq, const float* __restrict__ bk,
                            const float* __restrict__ bv, float* __restrict__ dst) {
  int i = blockIdx.x * blockDim.x + threadIdx.x;
  if (i >= LL * 1152) return;
  int l = i / 1152, n = i % 1152, which = n / EE, c = n % EE;
  const float* B = (which == 0) ? bq : ((which == 1) ? bk : bv);
  dst[i] = B[l * EE + c];
}

// ---------------------------------------------------------------------------
// Embedding: x[bt,:] = tok_emb[index[bt],:] + pos_emb[t,:]
// ---------------------------------------------------------------------------

__global__ __launch_bounds__(128) void k_embed(const int* __restrict__ idx,
                                               const float* __restrict__ tok,
                                               const float* __restrict__ pos,
                                               float* __restrict__ X) {
  int bt = blockIdx.x;
  int t = bt & (TT - 1);
  int v = idx[bt];
  for (int i = threadIdx.x; i < EE; i += 128)
    X[(size_t)bt * EE + i] = tok[(size_t)v * EE + i] + pos[(size_t)t * EE + i];
}

// ---------------------------------------------------------------------------
// LayerNorm (fp32 in, bf16 out)
// ---------------------------------------------------------------------------

__global__ __launch_bounds__(128) void k_ln(const float* __restrict__ X,
                                            const float* __restrict__ g,
                                            const float* __restrict__ bta,
                                            bf16_t* __restrict__ out) {
  __shared__ float s1[128], s2[128];
  int r = blockIdx.x, tid = threadIdx.x;
  const float* xr = X + (size_t)r * EE;
  float a = 0.f, b = 0.f;
  for (int i = tid; i < EE; i += 128) { float v = xr[i]; a += v; b += v * v; }
  s1[tid] = a; s2[tid] = b; __syncthreads();
  for (int s = 64; s > 0; s >>= 1) {
    if (tid < s) { s1[tid] += s1[tid + s]; s2[tid] += s2[tid + s]; }
    __syncthreads();
  }
  float mu  = s1[0] * (1.0f / EE);
  float var = s2[0] * (1.0f / EE) - mu * mu;
  float rstd = rsqrtf(var + 1e-5f);
  for (int i = tid; i < EE; i += 128)
    out[(size_t)r * EE + i] = (bf16_t)((xr[i] - mu) * rstd * g[i] + bta[i]);
}

// ---------------------------------------------------------------------------
// bf16 WMMA GEMM: C[M,N] = act(A[M,K] @ B^T + bias) (+ residual)
// A is [M,K] row-major; Bt is N-MAJOR [N,K] (pre-transposed weights).
// 128x64 tile / 256 threads (8 waves); each wave owns a 32x32 block
// (2x2 accumulators -> 4 WMMA per 32-wide K step from 2 A + 2 B fragments).
// Both LDS tiles match global layout -> ds_store_b128 only, no transpose.
// Requirements: M % 128 == 0, K % 32 == 0; N arbitrary (row-guarded).
// ---------------------------------------------------------------------------

__global__ __launch_bounds__(256) void k_gemm(const bf16_t* __restrict__ A,
                                              const bf16_t* __restrict__ Bt,
                                              const float* __restrict__ bias,
                                              const float* __restrict__ Res,
                                              float* __restrict__ Cout,
                                              bf16_t* __restrict__ Cbf,
                                              int M, int N, int K, int act) {
  __shared__ bf16_t As[128 * 40];  // [mrow][k], stride 40 keeps 16B row alignment
  __shared__ bf16_t Bs[64 * 40];   // [ncol][k]
  int tid  = threadIdx.x;
  int lane = tid & 31;
  int wave = tid >> 5;
  int m0 = blockIdx.y * 128;
  int n0 = blockIdx.x * 64;
  int mq = (wave & 3) * 32;        // wave's M offset within tile
  int nq = (wave >> 2) * 32;       // wave's N offset within tile

  f32x8 acc00 = zero8(), acc01 = zero8(), acc10 = zero8(), acc11 = zero8();

  int arow = tid >> 1, akc = (tid & 1) * 16;  // A loader: 128 rows x 2 chunks
  int nrow = tid >> 2, bkc = (tid & 3) * 8;   // B loader: 64 rows x 1 chunk
  const bool nvalid = (n0 + nrow < N);        // uniform per thread
  const bf16_t* arp = A + (size_t)(m0 + arow) * K + akc;
  const bf16_t* brp = Bt + (size_t)(nvalid ? (n0 + nrow) : 0) * K + bkc;

  for (int k0 = 0; k0 < K; k0 += 32) {
    bf16x8 av0 = *(const bf16x8*)(arp + k0);
    bf16x8 av1 = *(const bf16x8*)(arp + k0 + 8);
    bf16x8 bv  = nvalid ? *(const bf16x8*)(brp + k0) : zerob8();
    __syncthreads();
    *(bf16x8*)(As + arow * 40 + akc)     = av0;
    *(bf16x8*)(As + arow * 40 + akc + 8) = av1;
    *(bf16x8*)(Bs + nrow * 40 + bkc)     = bv;
    __syncthreads();

    // A fragments: lane<16 -> K {0..7,16..23}; lane>=16 -> K {8..15,24..31}
    Frag af0, af1, bf0, bf1;
    int am  = mq + (lane & 15);
    int akb = (lane < 16) ? 0 : 8;
    af0.h[0] = *(const bf16x8*)(As + am * 40 + akb);
    af0.h[1] = *(const bf16x8*)(As + am * 40 + 16 + akb);
    af1.h[0] = *(const bf16x8*)(As + (am + 16) * 40 + akb);
    af1.h[1] = *(const bf16x8*)(As + (am + 16) * 40 + 16 + akb);
    // B fragments: lane<16 -> K {0..7,16..23}; lane>=16 -> K {8..15,24..31}
    int bkb = (lane < 16) ? 0 : 8;
    int bn  = nq + (lane & 15);
    bf0.h[0] = *(const bf16x8*)(Bs + bn * 40 + bkb);
    bf0.h[1] = *(const bf16x8*)(Bs + bn * 40 + bkb + 16);
    bf1.h[0] = *(const bf16x8*)(Bs + (bn + 16) * 40 + bkb);
    bf1.h[1] = *(const bf16x8*)(Bs + (bn + 16) * 40 + bkb + 16);

    acc00 = wmma_bf16(af0.v, bf0.v, acc00);
    acc01 = wmma_bf16(af0.v, bf1.v, acc01);
    acc10 = wmma_bf16(af1.v, bf0.v, acc10);
    acc11 = wmma_bf16(af1.v, bf1.v, acc11);
  }

  // Epilogue: bias -> GELU -> residual -> store fp32 / bf16
  int col  = lane & 15;
  int roff = (lane < 16) ? 0 : 8;
#pragma unroll
  for (int sm = 0; sm < 2; ++sm) {
#pragma unroll
    for (int sn = 0; sn < 2; ++sn) {
#pragma unroll
      for (int r = 0; r < 8; ++r) {
        int grow = m0 + mq + sm * 16 + roff + r;
        int gcol = n0 + nq + sn * 16 + col;
        if (gcol < N) {
          float v;
          if (sm == 0) v = (sn == 0) ? acc00[r] : acc01[r];
          else         v = (sn == 0) ? acc10[r] : acc11[r];
          if (bias) v += bias[gcol];
          if (act == 1) v = 0.5f * v * (1.0f + erff(v * 0.70710678118f));
          if (Res) v += Res[(size_t)grow * N + gcol];
          if (Cout) Cout[(size_t)grow * N + gcol] = v;
          if (Cbf)  Cbf[(size_t)grow * N + gcol] = (bf16_t)v;
        }
      }
    }
  }
}

// ---------------------------------------------------------------------------
// QKV reshape: qkv[BT,1152] bf16 -> q/k [B,H,T,64] (d>=48 zero-padded),
// v -> TRANSPOSED [B,H,48,T] so attention P@V B-fragments are contiguous.
// ---------------------------------------------------------------------------

__global__ __launch_bounds__(64) void k_qkv_pad(const bf16_t* __restrict__ qkv,
                                                bf16_t* __restrict__ qp,
                                                bf16_t* __restrict__ kp,
                                                bf16_t* __restrict__ vpT) {
  int bid = blockIdx.x;                 // (b*H + h)*T + t
  int t = bid & (TT - 1);
  int h = (bid >> 9) & (HH - 1);
  int b = bid >> 12;
  int d = threadIdx.x;
  int bt = b * TT + t;
  size_t dsto = (size_t)bid * DP + d;
  bf16_t qv = (bf16_t)0.0f, kv = (bf16_t)0.0f;
  if (d < DD) {
    size_t so = (size_t)bt * 1152 + h * DD + d;
    qv = qkv[so];
    kv = qkv[so + EE];
    vpT[((size_t)(b * HH + h) * DD + d) * TT + t] = qkv[so + 2 * EE];
  }
  qp[dsto] = qv;
  kp[dsto] = kv;
}

// ---------------------------------------------------------------------------
// Causal attention, one wave32 per (b, h, 16-query tile).
// QK^T (WMMA, K=64 padded) -> fp32 scores in LDS -> in-wave softmax ->
// bf16 probs in LDS -> P@V (WMMA, causal extent rounded up to 32, zero pad).
// V is pre-transposed so its fragments are contiguous 16B chunks.
// ---------------------------------------------------------------------------

__global__ __launch_bounds__(32) void k_attn(const bf16_t* __restrict__ qp,
                                             const bf16_t* __restrict__ kp,
                                             const bf16_t* __restrict__ vpT,
                                             bf16_t* __restrict__ attnbf,
                                             float scale) {
  __shared__ float  sc[16 * 512];
  __shared__ bf16_t pb[16 * 512];
  int lane = threadIdx.x;
  int q0 = blockIdx.x * 16;
  int h  = blockIdx.y;
  int b  = blockIdx.z;
  size_t base  = ((size_t)(b * HH + h)) * TT * DP;   // q/k padded layout
  size_t vbase = ((size_t)(b * HH + h)) * DD * TT;   // v transposed layout
  int col  = lane & 15;
  int half = lane >> 4;

  // Q A-fragments for both 32-wide K steps (d 0..31, 32..63)
  Frag qf0, qf1;
  {
    const bf16_t* qrow = qp + base + (size_t)(q0 + col) * DP;
    int kb = half ? 8 : 0;
    qf0.h[0] = *(const bf16x8*)(qrow + kb);
    qf0.h[1] = *(const bf16x8*)(qrow + 16 + kb);
    qf1.h[0] = *(const bf16x8*)(qrow + 32 + kb);
    qf1.h[1] = *(const bf16x8*)(qrow + 48 + kb);
  }

  int ktmax = q0 >> 4;                    // inclusive: last key tile under causal mask
  for (int kt = 0; kt <= ktmax; ++kt) {
    const bf16_t* krow = kp + base + (size_t)(kt * 16 + col) * DP;
    int kb = half ? 8 : 0;
    Frag kf0, kf1;                        // B = K^T: element e -> d = step*32 + kb + e (e<8), step*32 + 16 + kb + (e-8) (e>=8)
    kf0.h[0] = *(const bf16x8*)(krow + kb);
    kf0.h[1] = *(const bf16x8*)(krow + kb + 16);
    kf1.h[0] = *(const bf16x8*)(krow + 32 + kb);
    kf1.h[1] = *(const bf16x8*)(krow + 32 + kb + 16);
    f32x8 acc = zero8();
    acc = wmma_bf16(qf0.v, kf0.v, acc);
    acc = wmma_bf16(qf1.v, kf1.v, acc);
    int kg = kt * 16 + col;
#pragma unroll
    for (int r = 0; r < 8; ++r) {
      int m  = r + (half ? 8 : 0);
      int qg = q0 + m;
      sc[m * 512 + kg] = (kg <= qg) ? acc[r] * scale : -INFINITY;
    }
  }
  __syncthreads();

  // Row softmax over keys [0, nvalid); 2 lanes per row (interleaved columns)
  int nvalid = q0 + 16;
  int m = col;
  float mx = -INFINITY;
  for (int n = half; n < nvalid; n += 2) mx = fmaxf(mx, sc[m * 512 + n]);
  mx = fmaxf(mx, __shfl_xor(mx, 16));
  float sum = 0.f;
  for (int n = half; n < nvalid; n += 2) {
    float e = expf(sc[m * 512 + n] - mx);
    sc[m * 512 + n] = e;
    sum += e;
  }
  sum += __shfl_xor(sum, 16);
  float rinv = 1.0f / sum;
  for (int n = half; n < nvalid; n += 2)
    pb[m * 512 + n] = (bf16_t)(sc[m * 512 + n] * rinv);
  int kk32 = (nvalid + 31) & ~31;
  if (kk32 > nvalid) {                    // zero-pad 16 key columns for K=32 WMMA
#pragma unroll
    for (int j = 0; j < 8; ++j)
      pb[m * 512 + nvalid + half * 8 + j] = (bf16_t)0.0f;
  }
  __syncthreads();

  // attn = P(16 x kk32) @ V(kk32 x 48); 3 N-tiles of 16
  f32x8 accv0 = zero8(), accv1 = zero8(), accv2 = zero8();
  int akb = half ? 8 : 0;
  int bkb = half ? 8 : 0;
  int nkt = kk32 >> 5;
  for (int kt = 0; kt < nkt; ++kt) {
    Frag pf;
    pf.h[0] = *(const bf16x8*)(pb + col * 512 + kt * 32 + akb);
    pf.h[1] = *(const bf16x8*)(pb + col * 512 + kt * 32 + 16 + akb);
#pragma unroll
    for (int nt = 0; nt < 3; ++nt) {
      const bf16_t* vrow = vpT + vbase + (size_t)(nt * 16 + col) * TT;
      Frag vf;                            // element e -> key = kt*32 + bkb + e (e<8), kt*32 + 16 + bkb + (e-8) (e>=8)
      vf.h[0] = *(const bf16x8*)(vrow + kt * 32 + bkb);
      vf.h[1] = *(const bf16x8*)(vrow + kt * 32 + bkb + 16);
      if (nt == 0)      accv0 = wmma_bf16(pf.v, vf.v, accv0);
      else if (nt == 1) accv1 = wmma_bf16(pf.v, vf.v, accv1);
      else              accv2 = wmma_bf16(pf.v, vf.v, accv2);
    }
  }

#pragma unroll
  for (int nt = 0; nt < 3; ++nt) {
#pragma unroll
    for (int r = 0; r < 8; ++r) {
      int row = q0 + r + (half ? 8 : 0);
      float v = (nt == 0) ? accv0[r] : ((nt == 1) ? accv1[r] : accv2[r]);
      attnbf[((size_t)(b * TT + row)) * EE + h * DD + nt * 16 + col] = (bf16_t)v;
    }
  }
}

// ---------------------------------------------------------------------------
// Loss: per-row log-softmax then mean reduce (no atomics -> graph-safe)
// ---------------------------------------------------------------------------

__global__ __launch_bounds__(256) void k_loss_row(const float* __restrict__ logits,
                                                  const int* __restrict__ tgt,
                                                  float* __restrict__ lpart) {
  __shared__ float red[256];
  int r = blockIdx.x, tid = threadIdx.x;
  const float* lr = logits + (size_t)r * VV;
  float mx = -INFINITY;
  for (int i = tid; i < VV; i += 256) mx = fmaxf(mx, lr[i]);
  red[tid] = mx; __syncthreads();
  for (int s = 128; s > 0; s >>= 1) {
    if (tid < s) red[tid] = fmaxf(red[tid], red[tid + s]);
    __syncthreads();
  }
  mx = red[0]; __syncthreads();
  float sum = 0.f;
  for (int i = tid; i < VV; i += 256) sum += expf(lr[i] - mx);
  red[tid] = sum; __syncthreads();
  for (int s = 128; s > 0; s >>= 1) {
    if (tid < s) red[tid] += red[tid + s];
    __syncthreads();
  }
  if (tid == 0) lpart[r] = -(lr[tgt[r]] - mx - logf(red[0]));
}

__global__ __launch_bounds__(256) void k_loss_final(const float* __restrict__ lpart,
                                                    float* __restrict__ loss) {
  __shared__ float red[256];
  int tid = threadIdx.x;
  float s = 0.f;
  for (int i = tid; i < BT; i += 256) s += lpart[i];
  red[tid] = s; __syncthreads();
  for (int st = 128; st > 0; st >>= 1) {
    if (tid < st) red[tid] += red[tid + st];
    __syncthreads();
  }
  if (tid == 0) *loss = red[0] / (float)BT;
}

// ---------------------------------------------------------------------------
// Host launcher
// ---------------------------------------------------------------------------

extern "C" void kernel_launch(void* const* d_in, const int* in_sizes, int n_in,
                              void* d_out, int out_size, void* d_ws, size_t ws_size,
                              hipStream_t stream) {
  const int*   idx   = (const int*)d_in[0];
  const int*   tgt   = (const int*)d_in[1];
  const float* tok   = (const float*)d_in[2];
  const float* pos   = (const float*)d_in[3];
  const float* Wq    = (const float*)d_in[4];
  const float* bq    = (const float*)d_in[5];
  const float* Wk    = (const float*)d_in[6];
  const float* bk    = (const float*)d_in[7];
  const float* Wv    = (const float*)d_in[8];
  const float* bv    = (const float*)d_in[9];
  const float* Wo    = (const float*)d_in[10];
  const float* bo    = (const float*)d_in[11];
  const float* ln1g  = (const float*)d_in[12];
  const float* ln1b  = (const float*)d_in[13];
  const float* ln2g  = (const float*)d_in[14];
  const float* ln2b  = (const float*)d_in[15];
  const float* W1    = (const float*)d_in[16];
  const float* b1    = (const float*)d_in[17];
  const float* W2    = (const float*)d_in[18];
  const float* b2    = (const float*)d_in[19];
  const float* lnfg  = (const float*)d_in[20];
  const float* lnfb  = (const float*)d_in[21];
  const float* lmb   = (const float*)d_in[22];

  char* ws = (char*)d_ws;
  size_t off = 0;
  auto alloc = [&](size_t bytes) -> char* {
    char* p = ws + off;
    off = (off + bytes + 255) & ~(size_t)255;
    return p;
  };

  float*  x      = (float*)alloc((size_t)BT * EE * 4);
  bf16_t* hbf    = (bf16_t*)alloc((size_t)BT * EE * 2);
  bf16_t* qkvbf  = (bf16_t*)alloc((size_t)BT * 1152 * 2);
  bf16_t* qp     = (bf16_t*)alloc((size_t)BB * HH * TT * DP * 2);
  bf16_t* kp     = (bf16_t*)alloc((size_t)BB * HH * TT * DP * 2);
  bf16_t* vpT    = (bf16_t*)alloc((size_t)BB * HH * DD * TT * 2);
  bf16_t* attnbf = (bf16_t*)alloc((size_t)BT * EE * 2);
  bf16_t* ffbf   = (bf16_t*)alloc((size_t)BT * FF_ * 2);
  bf16_t* qkvW   = (bf16_t*)alloc((size_t)LL * 1152 * EE * 2);  // [L][N=1152][K=384]
  float*  qkvB   = (float*)alloc((size_t)LL * 1152 * 4);
  bf16_t* WoBf   = (bf16_t*)alloc((size_t)LL * EE * EE * 2);    // [L][N][K]
  bf16_t* W1Bf   = (bf16_t*)alloc((size_t)LL * FF_ * EE * 2);   // [L][N=1536][K=384]
  bf16_t* W2Bf   = (bf16_t*)alloc((size_t)LL * EE * FF_ * 2);   // [L][N=384][K=1536]
  bf16_t* embBf  = (bf16_t*)alloc((size_t)VV * EE * 2);         // [N=V][K=E] (native)
  float*  lpart  = (float*)alloc((size_t)BT * 4);

  // --- weight prep: convert + transpose to N-major (per launch) ---
  {
    size_t n = (size_t)LL * 1152 * EE;
    k_pack_qkvw<<<(unsigned)((n + 255) / 256), 256, 0, stream>>>(Wq, Wk, Wv, qkvW);
    k_pack_qkvb<<<(LL * 1152 + 255) / 256, 256, 0, stream>>>(bq, bk, bv, qkvB);
    n = (size_t)LL * EE * EE;
    k_cvtT<<<(unsigned)((n + 255) / 256), 256, 0, stream>>>(Wo, WoBf, EE, EE, n);
    n = (size_t)LL * EE * FF_;
    k_cvtT<<<(unsigned)((n + 255) / 256), 256, 0, stream>>>(W1, W1Bf, EE, FF_, n);
    n = (size_t)LL * FF_ * EE;
    k_cvtT<<<(unsigned)((n + 255) / 256), 256, 0, stream>>>(W2, W2Bf, FF_, EE, n);
    n = (size_t)VV * EE;                         // tok_emb [V,E] is already N-major
    k_cvt<<<(unsigned)((n + 255) / 256), 256, 0, stream>>>(tok, embBf, n);
  }

  // --- embedding ---
  k_embed<<<BT, 128, 0, stream>>>(idx, tok, pos, x);

  const float scale = 0.14433756729740643f;  // 48^-0.5

  for (int l = 0; l < LL; ++l) {
    k_ln<<<BT, 128, 0, stream>>>(x, ln1g + l * EE, ln1b + l * EE, hbf);

    dim3 gQKV(1152 / 64, BT / 128);
    k_gemm<<<gQKV, 256, 0, stream>>>(hbf, qkvW + (size_t)l * 1152 * EE,
                                     qkvB + l * 1152, nullptr, nullptr, qkvbf,
                                     BT, 1152, EE, 0);

    k_qkv_pad<<<BB * HH * TT, 64, 0, stream>>>(qkvbf, qp, kp, vpT);

    dim3 gA(TT / 16, HH, BB);
    k_attn<<<gA, 32, 0, stream>>>(qp, kp, vpT, attnbf, scale);

    dim3 gO(EE / 64, BT / 128);
    k_gemm<<<gO, 256, 0, stream>>>(attnbf, WoBf + (size_t)l * EE * EE,
                                   bo + l * EE, x, x, nullptr,
                                   BT, EE, EE, 0);

    k_ln<<<BT, 128, 0, stream>>>(x, ln2g + l * EE, ln2b + l * EE, hbf);

    dim3 gF1(FF_ / 64, BT / 128);
    k_gemm<<<gF1, 256, 0, stream>>>(hbf, W1Bf + (size_t)l * FF_ * EE,
                                    b1 + l * FF_, nullptr, nullptr, ffbf,
                                    BT, FF_, EE, 1 /*gelu*/);

    dim3 gF2(EE / 64, BT / 128);
    k_gemm<<<gF2, 256, 0, stream>>>(ffbf, W2Bf + (size_t)l * EE * FF_,
                                    b2 + l * EE, x, x, nullptr,
                                    BT, EE, FF_, 0);
  }

  // --- final LN + tied logits ---
  k_ln<<<BT, 128, 0, stream>>>(x, lnfg, lnfb, hbf);

  float* logits = (float*)d_out;
  dim3 gL((VV + 63) / 64, BT / 128);
  k_gemm<<<gL, 256, 0, stream>>>(hbf, embBf, lmb, nullptr, logits, nullptr,
                                 BT, VV, EE, 0);

  // --- loss ---
  k_loss_row<<<BT, 256, 0, stream>>>(logits, tgt, lpart);
  k_loss_final<<<1, 256, 0, stream>>>(lpart, logits + (size_t)BT * VV);
}
